// GCN_20066087207444
// MI455X (gfx1250) — compile-verified, matrix operand order checked by reading, not run on hardware
//
#include <hip/hip_runtime.h>
#include <hip/hip_bf16.h>

// GCN forward on MI455X (gfx1250, wave32, WMMA + async-to-LDS staging).
// h1 = relu(norm_agg(x@W1)+b1); h2 = relu(norm_agg(h1@W2)+b2); out = h2@Wl + bl

typedef __attribute__((ext_vector_type(16))) _Float16 v16h;
typedef __attribute__((ext_vector_type(8)))  _Float16 v8h;
typedef __attribute__((ext_vector_type(8)))  float    v8f;
typedef __attribute__((ext_vector_type(4)))  int      v4i;

#define HDIM 512
#define ODIM 16
#define LDS_STRIDE 40   // halves per staged column: 32 data + 8 pad (80B) -> bank-conflict-free

#if defined(__has_builtin)
#if __has_builtin(__builtin_amdgcn_global_load_async_to_lds_b128)
#define USE_ASYNC_LDS 1
#endif
#endif

// ---- fragment helpers -------------------------------------------------------
// 16-bit A/B fragment per ISA layout: lane l holds row/col (l&15);
// halves 0..7 = K kb..kb+7, halves 8..15 = K kb+16..kb+23, kb=(l>>4)*8.
__device__ __forceinline__ v16h load_frag16(const _Float16* __restrict__ p) {
  v8h lo = *(const v8h*)(p);
  v8h hi = *(const v8h*)(p + 16);
  return __builtin_shufflevector(lo, hi, 0,1,2,3,4,5,6,7,8,9,10,11,12,13,14,15);
}

__device__ __forceinline__ v8f wmma_f16(v16h a, v16h b, v8f c) {
  // (neg_a, A, neg_b, B, c_mod, C, reuse_a, reuse_b)
  return __builtin_amdgcn_wmma_f32_16x16x32_f16(false, a, false, b, (short)0, c,
                                                false, false);
}

// ---- prep kernels -----------------------------------------------------------
__global__ void k_zero_f32(float* __restrict__ p, int n) {
  int i = blockIdx.x * blockDim.x + threadIdx.x;
  if (i < n) p[i] = 0.0f;
}

__global__ void k_deg(const int* __restrict__ ei, float* __restrict__ deg, int E) {
  int e = blockIdx.x * blockDim.x + threadIdx.x;
  if (e < E) atomicAdd(&deg[ei[E + e]], 1.0f);   // dst = ei[1][e]
}

__global__ void k_dinv(float* __restrict__ d, int n) {
  int i = blockIdx.x * blockDim.x + threadIdx.x;
  if (i < n) d[i] = rsqrtf(d[i] + 1.0f);         // self-loop: deg >= 1
}

__global__ void k_f32_to_f16(const float* __restrict__ s, _Float16* __restrict__ d, int n) {
  int i = blockIdx.x * blockDim.x + threadIdx.x;
  if (i < n) d[i] = (_Float16)s[i];
}

// Wt[n*K + k] = (f16) W[k*Ncol + n]  -- B^T so B-fragments load contiguously
__global__ void k_transpose_f16(const float* __restrict__ W, _Float16* __restrict__ Wt,
                                int K, int Ncol) {
  int i = blockIdx.x * blockDim.x + threadIdx.x;
  if (i >= K * Ncol) return;
  int k = i / Ncol, n = i - k * Ncol;
  Wt[(size_t)n * K + k] = (_Float16)W[i];
}

// ---- tiled WMMA GEMM: C[M x 512] = A[M x 512] * B, Bt = B^T [512 x 512] -----
// Block (8 waves) owns a 128x64 tile. B strip chunks (64 cols x 32 K = 4KB) are
// staged in LDS, double-buffered, via async global->LDS copies; all 8 waves
// share each chunk. A fragments are register-pipelined one K-step ahead.
__global__ void __launch_bounds__(256) k_gemm_tiled(
    const _Float16* __restrict__ A, const _Float16* __restrict__ Bt,
    float* __restrict__ C, int M) {
  __shared__ _Float16 sB[2][64 * LDS_STRIDE];
  const int K = HDIM;
  const int lane = threadIdx.x & 31;
  const int wave = threadIdx.x >> 5;
  const int mBlock = blockIdx.x >> 3;        // 8 n-strips per m-block
  const int nBase  = (blockIdx.x & 7) * 64;
  const int r  = lane & 15;
  const int kb = (lane >> 4) << 3;

  int rowL = mBlock * 128 + wave * 16 + r;   // clamped for loads; stores guarded
  if (rowL >= M) rowL = M - 1;
  const _Float16* aP = A + (size_t)rowL * K + kb;

  // staging assignment: thread t moves one 16B segment: col = t>>2, seg = t&3
  const int tcol = threadIdx.x >> 2;
  const int tseg = threadIdx.x & 3;
  const _Float16* gB = Bt + (size_t)(nBase + tcol) * K + tseg * 8;
  _Float16* lB = &sB[0][tcol * LDS_STRIDE + tseg * 8];
  const int bufOff = 64 * LDS_STRIDE;

  auto issue = [&](int ks, int buf) {
#if USE_ASYNC_LDS
    __builtin_amdgcn_global_load_async_to_lds_b128(
        (__attribute__((address_space(1))) v4i*)(gB + ks),
        (__attribute__((address_space(3))) v4i*)(lB + buf * bufOff), 0, 0);
#else
    *(v8h*)(lB + buf * bufOff) = *(const v8h*)(gB + ks);
#endif
  };

  issue(0, 0);
  v16h a = load_frag16(aP);
  v8f c0 = {}, c1 = {}, c2 = {}, c3 = {};
  int cur = 0;
  for (int ks = 0; ks < K; ks += 32) {
    const bool more = (ks + 32) < K;
    if (more) issue(ks + 32, cur ^ 1);       // prefetch next chunk into other buffer
#if USE_ASYNC_LDS
    if (more) asm volatile("s_wait_asynccnt 0x1" ::: "memory");
    else      asm volatile("s_wait_asynccnt 0x0" ::: "memory");
#endif
    __syncthreads();                          // chunk(ks) visible to all waves

    v16h aN = a;
    if (more) aN = load_frag16(aP + ks + 32); // register-pipelined A prefetch

    const _Float16* lb = &sB[cur][0];
    v16h b0 = load_frag16(lb + (r +  0) * LDS_STRIDE + kb);
    v16h b1 = load_frag16(lb + (r + 16) * LDS_STRIDE + kb);
    v16h b2 = load_frag16(lb + (r + 32) * LDS_STRIDE + kb);
    v16h b3 = load_frag16(lb + (r + 48) * LDS_STRIDE + kb);
    c0 = wmma_f16(a, b0, c0);
    c1 = wmma_f16(a, b1, c1);
    c2 = wmma_f16(a, b2, c2);
    c3 = wmma_f16(a, b3, c3);

    __syncthreads();                          // done reading buf[cur]; next iter may overwrite
    a = aN;
    cur ^= 1;
  }

  // D layout: VGPR v -> row = base + v + 8*(lane>>4), col = nBase + (lane&15)
  const int rowBase = mBlock * 128 + wave * 16 + ((lane >> 4) << 3);
  const int col = nBase + (lane & 15);
#pragma unroll
  for (int v = 0; v < 8; ++v) {
    int rr = rowBase + v;
    if (rr < M) {
      float* cp = C + (size_t)rr * HDIM + col;
      cp[ 0] = c0[v];
      cp[16] = c1[v];
      cp[32] = c2[v];
      cp[48] = c3[v];
    }
  }
}

// ---- final classifier GEMM: out[M x 16] = A[M x 512] * Wl + bl --------------
__global__ void __launch_bounds__(256) k_gemm_out(
    const _Float16* __restrict__ A, const _Float16* __restrict__ Bt,
    const float* __restrict__ bias, float* __restrict__ C, int M) {
  const int K = HDIM;
  const int lane = threadIdx.x & 31;
  const int mTile = blockIdx.x * (blockDim.x >> 5) + (threadIdx.x >> 5);
  if (mTile * 16 >= M) return;
  const int r  = lane & 15;
  const int kb = (lane >> 4) << 3;
  const _Float16* aP = A  + (size_t)(mTile * 16 + r) * K + kb;
  const _Float16* bP = Bt + (size_t)r * K + kb;      // 16 output cols total
  v8f c0 = {};
  for (int ks = 0; ks < K; ks += 32) {
    v16h a = load_frag16(aP + ks);
    v16h b = load_frag16(bP + ks);
    c0 = wmma_f16(a, b, c0);
  }
  const int rowBase = mTile * 16 + ((lane >> 4) << 3);
  const int col = lane & 15;
  const float bv = bias[col];
#pragma unroll
  for (int v = 0; v < 8; ++v)
    C[(size_t)(rowBase + v) * ODIM + col] = c0[v] + bv;
}

// ---- aggregation ------------------------------------------------------------
__global__ void k_agg_self(const float* __restrict__ hx, const float* __restrict__ dinv,
                           float* __restrict__ agg, int total) {
  int i = blockIdx.x * blockDim.x + threadIdx.x;
  if (i >= total) return;
  float d = dinv[i >> 9];                 // HDIM == 512
  agg[i] = d * d * hx[i];
}

__global__ void k_agg_edges(const int* __restrict__ ei, const float* __restrict__ dinv,
                            const float* __restrict__ hx, float* __restrict__ agg, int E) {
  int t = blockIdx.x * blockDim.x + threadIdx.x;
  int e = t >> 7;                         // 128 chunks of 4 floats per edge
  if (e >= E) return;
  int f = (t & 127) << 2;
  int s = ei[e], d = ei[E + e];
  float norm = dinv[s] * dinv[d];
  const float4 v = *(const float4*)(hx + (size_t)s * HDIM + f);
  float* ap = agg + (size_t)d * HDIM + f;
  atomicAdd(ap + 0, norm * v.x);
  atomicAdd(ap + 1, norm * v.y);
  atomicAdd(ap + 2, norm * v.z);
  atomicAdd(ap + 3, norm * v.w);
}

__global__ void k_bias_relu_f16(const float* __restrict__ agg, const float* __restrict__ b,
                                _Float16* __restrict__ outh, int total) {
  int i = blockIdx.x * blockDim.x + threadIdx.x;
  if (i >= total) return;
  float v = agg[i] + b[i & (HDIM - 1)];
  outh[i] = (_Float16)(v > 0.0f ? v : 0.0f);
}

// ---- host launch ------------------------------------------------------------
extern "C" void kernel_launch(void* const* d_in, const int* in_sizes, int n_in,
                              void* d_out, int out_size, void* d_ws, size_t ws_size,
                              hipStream_t stream) {
  const float* x  = (const float*)d_in[0];
  const int*   ei = (const int*)  d_in[1];
  const float* W1 = (const float*)d_in[2];
  const float* b1 = (const float*)d_in[3];
  const float* W2 = (const float*)d_in[4];
  const float* b2 = (const float*)d_in[5];
  const float* Wl = (const float*)d_in[6];
  const float* bl = (const float*)d_in[7];
  float* out = (float*)d_out;

  const int N = in_sizes[0] / HDIM;   // 10000
  const int E = in_sizes[1] / 2;      // 160000
  const int totNH = N * HDIM;

  // carve workspace (all offsets 256B aligned)
  char* w = (char*)d_ws;
  auto carve = [&](size_t bytes) -> char* {
    char* p = w; w += (bytes + 255) & ~(size_t)255; return p;
  };
  float*    dinv = (float*)   carve(sizeof(float) * N);
  _Float16* Ah   = (_Float16*)carve(sizeof(_Float16) * (size_t)totNH);
  _Float16* W1t  = (_Float16*)carve(sizeof(_Float16) * HDIM * HDIM);
  _Float16* W2t  = (_Float16*)carve(sizeof(_Float16) * HDIM * HDIM);
  _Float16* Wlt  = (_Float16*)carve(sizeof(_Float16) * HDIM * ODIM);
  float*    hx   = (float*)   carve(sizeof(float) * (size_t)totNH);
  float*    agg  = (float*)   carve(sizeof(float) * (size_t)totNH);

  const int B = 256;
  // degree -> dinv
  k_zero_f32<<<(N + B - 1) / B, B, 0, stream>>>(dinv, N);
  k_deg<<<(E + B - 1) / B, B, 0, stream>>>(ei, dinv, E);
  k_dinv<<<(N + B - 1) / B, B, 0, stream>>>(dinv, N);
  // precision/layout prep
  k_f32_to_f16<<<(totNH + B - 1) / B, B, 0, stream>>>(x, Ah, totNH);
  k_transpose_f16<<<(HDIM * HDIM + B - 1) / B, B, 0, stream>>>(W1, W1t, HDIM, HDIM);
  k_transpose_f16<<<(HDIM * HDIM + B - 1) / B, B, 0, stream>>>(W2, W2t, HDIM, HDIM);
  k_transpose_f16<<<(HDIM * ODIM + B - 1) / B, B, 0, stream>>>(Wl, Wlt, HDIM, ODIM);

  const int mBlocks = (N + 127) / 128;              // 79 (128 rows per block)
  const int gemmGrid = mBlocks * 8;                 // x8 n-strips of 64 cols
  const int mTiles = (N + 15) / 16;                 // 625

  for (int layer = 0; layer < 2; ++layer) {
    const _Float16* Wt = (layer == 0) ? W1t : W2t;
    const float*    bb = (layer == 0) ? b1  : b2;
    k_gemm_tiled<<<gemmGrid, B, 0, stream>>>(Ah, Wt, hx, N);
    k_agg_self<<<(totNH + B - 1) / B, B, 0, stream>>>(hx, dinv, agg, totNH);
    k_agg_edges<<<(E * 128 + B - 1) / B, B, 0, stream>>>(ei, dinv, hx, agg, E);
    k_bias_relu_f16<<<(totNH + B - 1) / B, B, 0, stream>>>(agg, bb, Ah, totNH);
  }
  k_gemm_out<<<(mTiles + 7) / 8, B, 0, stream>>>(Ah, Wlt, bl, out, N);
}
